// GRU_37349035606699
// MI455X (gfx1250) — compile-verified, matrix operand order checked by reading, not run on hardware
//
#include <hip/hip_runtime.h>
#include <hip/hip_bf16.h>

#define B_SZ   256
#define T_SZ   512
#define IN_D   256
#define H_D    1024
#define OUT_D  256

// LDS row strides (elements), padded: stride*2B must be 16B-multiple and
// (stride/2) mod 64 != 0 so successive rows land on shifted banks.
#define AX_STRIDE (IN_D + 8)   // 264 elems = 528 B  (132 dw, =4 mod 64)
#define AH_STRIDE (H_D + 8)    // 1032 elems = 2064 B (516 dw, =4 mod 64)

typedef __attribute__((ext_vector_type(16))) __bf16 v16bf;
typedef __attribute__((ext_vector_type(8)))  __bf16 v8bf;
typedef __attribute__((ext_vector_type(8)))  float  v8f;
typedef int v4i __attribute__((vector_size(16)));
typedef __attribute__((address_space(1))) v4i* v4i_gptr;
typedef __attribute__((address_space(3))) v4i* v4i_lptr;

union V16U { v16bf v; struct { v8bf lo, hi; } s; };

#if defined(__has_builtin)
#if __has_builtin(__builtin_amdgcn_global_load_async_to_lds_b128)
#define HAVE_ASYNC_LDS 1
#endif
#endif

// One 16-byte Global->LDS transfer (async path if the toolchain exposes it).
__device__ __forceinline__ void g2l_b128(const __bf16* __restrict__ g, __bf16* l) {
#ifdef HAVE_ASYNC_LDS
    __builtin_amdgcn_global_load_async_to_lds_b128(
        (v4i_gptr)(g), (v4i_lptr)(l), 0, 0);
#else
    *reinterpret_cast<v8bf*>(l) = *reinterpret_cast<const v8bf*>(g);
#endif
}

__device__ __forceinline__ void g2l_commit_and_barrier() {
#ifdef HAVE_ASYNC_LDS
#if __has_builtin(__builtin_amdgcn_s_wait_asynccnt)
    __builtin_amdgcn_s_wait_asynccnt(0);
#else
    asm volatile("s_wait_asynccnt 0x0" ::: "memory");
#endif
#endif
    __syncthreads();
}

// Load one wave32 A/B fragment of a 16x32 bf16 tile (row-major, row = lane%16).
// ISA layout: lanes 0-15 hold K=[k0..k0+7] in v0-3 and K=[k0+16..k0+23] in v4-7;
// lanes 16-31 hold the +8 halves. 'base' already includes row*stride.
__device__ __forceinline__ v16bf ld_frag(const __bf16* base, int k0, int khalf) {
    V16U u;
    u.s.lo = *reinterpret_cast<const v8bf*>(base + k0 + khalf * 8);
    u.s.hi = *reinterpret_cast<const v8bf*>(base + k0 + 16 + khalf * 8);
    return u.v;
}

__device__ __forceinline__ v8f bf16_wmma(v16bf a, v16bf b, v8f c) {
    return __builtin_amdgcn_wmma_f32_16x16x32_bf16(false, a, false, b, (short)0, c, false, false);
}

__device__ __forceinline__ float sigmoid_f(float x) {
    return 1.0f / (1.0f + __expf(-x));
}

// ---------------- utility kernels ----------------

__global__ void f32_to_bf16_k(const float* __restrict__ in, __bf16* __restrict__ out, int n) {
    int i = blockIdx.x * blockDim.x + threadIdx.x;
    if (i < n) out[i] = (__bf16)in[i];
}

__global__ void init_h_k(float* __restrict__ hf, __bf16* __restrict__ hb, int n) {
    int i = blockIdx.x * blockDim.x + threadIdx.x;
    if (i < n) { hf[i] = 0.0f; hb[i] = (__bf16)0.0f; }
}

// ---------------- GRU timestep ----------------
// Block = 8 waves sharing one M-block (32 batch rows). The A panel
// (32 x 256 of x_t, 32 x 1024 of h_prev, bf16) is staged once per block into
// LDS via async Global->LDS, then every wave reads its A fragments from LDS
// (8x traffic cut on A). B (weight rows) streams from the 192MB L2.
// Each wave owns a 32(M) x 16(N=hidden) patch: 2 M-tiles x 4 gate accumulators
// (r, z combined x+h; xn x-only; hn h-only).
// grid = 64 blocks * 8 waves = 8 M-blocks * 64 N-blocks.
__global__ __launch_bounds__(256) void gru_step_k(
    const __bf16* __restrict__ xb_t,   // x bf16 at timestep t; batch-row stride T*IN
    const __bf16* __restrict__ hb_in,  // [B, H] bf16
    const float*  __restrict__ hf_in,  // [B, H] f32
    const __bf16* __restrict__ Wxb,    // [3H, IN] bf16
    const __bf16* __restrict__ Whb,    // [3H, H]  bf16
    const float*  __restrict__ bx,     // [3H]
    const float*  __restrict__ bh,     // [3H]
    __bf16* __restrict__ hb_out,       // [B, H]
    float*  __restrict__ hf_out)       // [B, H]
{
    __shared__ __bf16 Ax[32 * AX_STRIDE];  // 16.5 KB
    __shared__ __bf16 Ah[32 * AH_STRIDE];  // 64.5 KB

    const int tid   = threadIdx.x;
    const int lane  = tid & 31;
    const int wave  = tid >> 5;
    const int mblk  = blockIdx.x >> 3;                    // 0..7  (32 rows each)
    const int nblk  = ((blockIdx.x & 7) << 3) + wave;     // 0..63 (16 cols each)
    const int jbase = nblk * 16;
    const int mbase = mblk * 32;

    // ---- stage A panel into LDS (async, 16B granules) ----
    // x_t: 32 rows x 256 elems = 1024 chunks (32 chunks/row)
    #pragma unroll
    for (int it = 0; it < 4; ++it) {
        int c   = tid + it * 256;
        int row = c >> 5, kc = c & 31;
        g2l_b128(xb_t + (size_t)(mbase + row) * (T_SZ * IN_D) + kc * 8,
                 &Ax[row * AX_STRIDE + kc * 8]);
    }
    // h_prev: 32 rows x 1024 elems = 4096 chunks (128 chunks/row)
    #pragma unroll
    for (int it = 0; it < 16; ++it) {
        int c   = tid + it * 256;
        int row = c >> 7, kc = c & 127;
        g2l_b128(hb_in + (size_t)(mbase + row) * H_D + kc * 8,
                 &Ah[row * AH_STRIDE + kc * 8]);
    }
    g2l_commit_and_barrier();

    const int mloc  = lane & 15;   // row within 16-row tile
    const int khalf = lane >> 4;   // which K-half this lane holds

    v8f acc_r[2]  = {v8f{}, v8f{}};
    v8f acc_z[2]  = {v8f{}, v8f{}};
    v8f acc_xn[2] = {v8f{}, v8f{}};
    v8f acc_hn[2] = {v8f{}, v8f{}};

    // A fragment bases in LDS
    const __bf16* ax0 = &Ax[(0 * 16 + mloc) * AX_STRIDE];
    const __bf16* ax1 = &Ax[(1 * 16 + mloc) * AX_STRIDE];
    const __bf16* ah0 = &Ah[(0 * 16 + mloc) * AH_STRIDE];
    const __bf16* ah1 = &Ah[(1 * 16 + mloc) * AH_STRIDE];
    // B (weight-row) bases in global; W stored [N,K] row-major == column-major B
    const __bf16* wxr = Wxb + (size_t)(0 * H_D + jbase + mloc) * IN_D;
    const __bf16* wxz = Wxb + (size_t)(1 * H_D + jbase + mloc) * IN_D;
    const __bf16* wxn = Wxb + (size_t)(2 * H_D + jbase + mloc) * IN_D;
    const __bf16* whr = Whb + (size_t)(0 * H_D + jbase + mloc) * H_D;
    const __bf16* whz = Whb + (size_t)(1 * H_D + jbase + mloc) * H_D;
    const __bf16* whn = Whb + (size_t)(2 * H_D + jbase + mloc) * H_D;

    // ---- phase 1: x_t @ Wx^T, K = 256 ----
    #pragma unroll 2
    for (int k0 = 0; k0 < IN_D; k0 += 32) {
        v16bf a0 = ld_frag(ax0, k0, khalf);
        v16bf a1 = ld_frag(ax1, k0, khalf);
        v16bf br = ld_frag(wxr, k0, khalf);
        v16bf bz = ld_frag(wxz, k0, khalf);
        v16bf bn = ld_frag(wxn, k0, khalf);
        acc_r[0]  = bf16_wmma(a0, br, acc_r[0]);  acc_r[1]  = bf16_wmma(a1, br, acc_r[1]);
        acc_z[0]  = bf16_wmma(a0, bz, acc_z[0]);  acc_z[1]  = bf16_wmma(a1, bz, acc_z[1]);
        acc_xn[0] = bf16_wmma(a0, bn, acc_xn[0]); acc_xn[1] = bf16_wmma(a1, bn, acc_xn[1]);
    }

    // ---- phase 2: h_prev @ Wh^T, K = 1024 ----
    #pragma unroll 2
    for (int k0 = 0; k0 < H_D; k0 += 32) {
        v16bf a0 = ld_frag(ah0, k0, khalf);
        v16bf a1 = ld_frag(ah1, k0, khalf);
        v16bf br = ld_frag(whr, k0, khalf);
        v16bf bz = ld_frag(whz, k0, khalf);
        v16bf bn = ld_frag(whn, k0, khalf);
        acc_r[0]  = bf16_wmma(a0, br, acc_r[0]);  acc_r[1]  = bf16_wmma(a1, br, acc_r[1]);
        acc_z[0]  = bf16_wmma(a0, bz, acc_z[0]);  acc_z[1]  = bf16_wmma(a1, bz, acc_z[1]);
        acc_hn[0] = bf16_wmma(a0, bn, acc_hn[0]); acc_hn[1] = bf16_wmma(a1, bn, acc_hn[1]);
    }

    // ---- elementwise gates + state update ----
    // C/D layout: VGPR i holds M = i + 8*(lane>=16), N = lane%16.
    const int j  = jbase + (lane & 15);
    const int mo = (lane >> 4) * 8;
    const float bxr = bx[j],            bhr = bh[j];
    const float bxz = bx[H_D + j],      bhz = bh[H_D + j];
    const float bxn = bx[2 * H_D + j],  bhn = bh[2 * H_D + j];

    #pragma unroll
    for (int mt = 0; mt < 2; ++mt) {
        #pragma unroll
        for (int i = 0; i < 8; ++i) {
            const int b = mbase + mt * 16 + mo + i;
            float r  = sigmoid_f(acc_r[mt][i] + bxr + bhr);
            float z  = sigmoid_f(acc_z[mt][i] + bxz + bhz);
            float n  = tanhf((acc_xn[mt][i] + bxn) + r * (acc_hn[mt][i] + bhn));
            float hp = hf_in[(size_t)b * H_D + j];
            float h  = (1.0f - z) * hp + z * n;   // nonstandard blend, follows reference
            hf_out[(size_t)b * H_D + j] = h;
            hb_out[(size_t)b * H_D + j] = (__bf16)h;
        }
    }
}

// ---------------- final projection: out = h @ Wf^T + bf ----------------
__global__ __launch_bounds__(256) void out_proj_k(
    const __bf16* __restrict__ hb,    // [B, H] bf16 (final h)
    const __bf16* __restrict__ Wfb,   // [OUT, H] bf16
    const float*  __restrict__ bfv,   // [OUT]
    float* __restrict__ out)          // [B, OUT]
{
    const int lane  = threadIdx.x & 31;
    const int wave  = threadIdx.x >> 5;
    const int task  = blockIdx.x * 8 + wave;   // 0..255
    const int nblk  = task & 15;
    const int mblk  = task >> 4;
    const int mloc  = lane & 15;
    const int khalf = lane >> 4;

    const __bf16* ap = hb  + (size_t)(mblk * 16 + mloc) * H_D;
    const __bf16* bp = Wfb + (size_t)(nblk * 16 + mloc) * H_D;

    v8f acc = v8f{};
    #pragma unroll 4
    for (int k0 = 0; k0 < H_D; k0 += 32) {
        v16bf a = ld_frag(ap, k0, khalf);
        v16bf b = ld_frag(bp, k0, khalf);
        acc = bf16_wmma(a, b, acc);
    }

    const int n  = nblk * 16 + (lane & 15);
    const int mo = (lane >> 4) * 8;
    const float bias = bfv[n];
    #pragma unroll
    for (int i = 0; i < 8; ++i) {
        const int m = mblk * 16 + mo + i;
        out[(size_t)m * OUT_D + n] = acc[i] + bias;
    }
}

// ---------------- launcher ----------------

extern "C" void kernel_launch(void* const* d_in, const int* in_sizes, int n_in,
                              void* d_out, int out_size, void* d_ws, size_t ws_size,
                              hipStream_t stream) {
    const float* x   = (const float*)d_in[0];
    const float* Wx  = (const float*)d_in[1];
    const float* bx  = (const float*)d_in[2];
    const float* Wh  = (const float*)d_in[3];
    const float* bh  = (const float*)d_in[4];
    const float* Wf  = (const float*)d_in[5];
    const float* bfv = (const float*)d_in[6];
    float* out = (float*)d_out;

    char* ws = (char*)d_ws;
    size_t off = 0;
    auto carve = [&](size_t bytes) {
        void* p = ws + off;
        off += (bytes + 255) & ~(size_t)255;
        return p;
    };
    const size_t NX  = (size_t)B_SZ * T_SZ * IN_D;
    const size_t NWX = (size_t)3 * H_D * IN_D;
    const size_t NWH = (size_t)3 * H_D * H_D;
    const size_t NWF = (size_t)OUT_D * H_D;
    const size_t NH  = (size_t)B_SZ * H_D;

    __bf16* xb  = (__bf16*)carve(NX  * sizeof(__bf16));
    __bf16* Wxb = (__bf16*)carve(NWX * sizeof(__bf16));
    __bf16* Whb = (__bf16*)carve(NWH * sizeof(__bf16));
    __bf16* Wfb = (__bf16*)carve(NWF * sizeof(__bf16));
    float*  hf0 = (float*)carve(NH * sizeof(float));
    float*  hf1 = (float*)carve(NH * sizeof(float));
    __bf16* hb0 = (__bf16*)carve(NH * sizeof(__bf16));
    __bf16* hb1 = (__bf16*)carve(NH * sizeof(__bf16));

    f32_to_bf16_k<<<(int)((NX  + 255) / 256), 256, 0, stream>>>(x,  xb,  (int)NX);
    f32_to_bf16_k<<<(int)((NWX + 255) / 256), 256, 0, stream>>>(Wx, Wxb, (int)NWX);
    f32_to_bf16_k<<<(int)((NWH + 255) / 256), 256, 0, stream>>>(Wh, Whb, (int)NWH);
    f32_to_bf16_k<<<(int)((NWF + 255) / 256), 256, 0, stream>>>(Wf, Wfb, (int)NWF);
    init_h_k<<<(int)((NH + 255) / 256), 256, 0, stream>>>(hf0, hb0, (int)NH);

    // serial recurrence: one fused WMMA kernel per timestep, ping-pong h buffers
    for (int t = 0; t < T_SZ; ++t) {
        const __bf16* hbi = (t & 1) ? hb1 : hb0;
        const float*  hfi = (t & 1) ? hf1 : hf0;
        __bf16* hbo = (t & 1) ? hb0 : hb1;
        float*  hfo = (t & 1) ? hf0 : hf1;
        gru_step_k<<<64, 256, 0, stream>>>(xb + (size_t)t * IN_D,
                                           hbi, hfi, Wxb, Whb, bx, bh, hbo, hfo);
    }

    // after t = 511 (odd), final h lives in buffer 0
    out_proj_k<<<32, 256, 0, stream>>>(hb0, Wfb, bfv, out);
}
